// Attention_83580063580522
// MI455X (gfx1250) — compile-verified, matrix operand order checked by reading, not run on hardware
//
#include <hip/hip_runtime.h>

// ---------------- problem constants ----------------
#define Bsz 2
#define Ssz 2048
#define DM  2048
#define Hq  32
#define KVh 8
#define HD  64
#define BS  (Bsz * Ssz)       // 4096
#define NQKV 3072             // 2048 q + 512 k + 512 v
#define HHD (Hq * HD)         // 2048
#define NEG_BIG (-1.0e30f)

typedef __bf16 v16bf __attribute__((ext_vector_type(16)));
typedef __bf16 bf2   __attribute__((ext_vector_type(2)));
typedef float  v8f   __attribute__((ext_vector_type(8)));

// round-to-nearest-even fp32 -> bf16
__device__ inline __bf16 f2bf(float f) {
    unsigned u = __builtin_bit_cast(unsigned, f);
    unsigned r = (u + 0x7FFFu + ((u >> 16) & 1u)) >> 16;
    return __builtin_bit_cast(__bf16, (unsigned short)r);
}

// Load one 16x32 bf16 A/B fragment for v_wmma_f32_16x16x32_bf16.
// p points at element [row][k=0]; K runs contiguously in memory.
// Pair j holds K = ((j&4)?16:0) + 8*half + {2*(j&3), 2*(j&3)+1}   (ISA 7.12.2)
__device__ inline v16bf load_frag16(const __bf16* __restrict__ p, int half) {
    const unsigned* p32 = (const unsigned*)p;
    v16bf r;
#pragma unroll
    for (int j = 0; j < 8; ++j) {
        int k0 = ((j & 4) ? 16 : 0) + half * 8 + 2 * (j & 3);
        bf2 q = __builtin_bit_cast(bf2, p32[k0 >> 1]);
        r[2 * j]     = q.x;
        r[2 * j + 1] = q.y;
    }
    return r;
}

__device__ inline v8f wmma_bf16(v16bf a, v16bf b, v8f c) {
    return __builtin_amdgcn_wmma_f32_16x16x32_bf16(
        /*neg_a=*/false, a, /*neg_b=*/false, b,
        /*c_mod=*/(short)0, c, /*reuse_a=*/false, /*reuse_b=*/false);
}

// ---------------- elementwise convert ----------------
__global__ __launch_bounds__(256) void k_conv_bf16(const float* __restrict__ src,
                                                   __bf16* __restrict__ dst, int n) {
    int i = blockIdx.x * blockDim.x + threadIdx.x;
    if (i < n) dst[i] = f2bf(src[i]);
}

// src[Kdim][Ndim] fp32  ->  dst[Ndim][Kdim] bf16  (B-operand wants contiguous K)
__global__ __launch_bounds__(256) void k_transpose_bf16(const float* __restrict__ src,
                                                        __bf16* __restrict__ dst,
                                                        int Kdim, int Ndim) {
    int i = blockIdx.x * blockDim.x + threadIdx.x;
    if (i >= Kdim * Ndim) return;
    int n = i / Kdim, k = i % Kdim;
    dst[i] = f2bf(src[(size_t)k * Ndim + n]);
}

// ---------------- bf16 WMMA GEMM:  C[M][N] = A[M][K] * Bt[N][K]^T + bias ----------------
// wave tile 32(M) x 64(N): 8 WMMAs per 32-deep K-step sharing 4 B fragments.
// All 6 fragment loads issued before the WMMA burst so loadcnt waits are partial.
// block = 8 waves -> 256(M) x 64(N)
__global__ __launch_bounds__(256) void k_gemm_bf16(const __bf16* __restrict__ A,
                                                   const __bf16* __restrict__ Bt,
                                                   float* __restrict__ C,
                                                   const float* __restrict__ bias,
                                                   int M, int N, int K, int bias_from) {
    const int lane = threadIdx.x & 31;
    const int wave = threadIdx.x >> 5;
    const int half = lane >> 4;
    const int ln   = lane & 15;
    const int m0 = (blockIdx.y * 8 + wave) * 32;
    const int n0 = blockIdx.x * 64;
    if (m0 >= M) return;

    v8f acc0[4] = {};
    v8f acc1[4] = {};
    const __bf16* Arow0 = A + (size_t)(m0 + ln) * K;
    const __bf16* Arow1 = A + (size_t)(m0 + 16 + ln) * K;
    for (int k0 = 0; k0 < K; k0 += 32) {
        // issue all loads for this K-step first (independent live ranges)
        v16bf a0 = load_frag16(Arow0 + k0, half);
        v16bf a1 = load_frag16(Arow1 + k0, half);
        v16bf bfr[4];
#pragma unroll
        for (int t = 0; t < 4; ++t) {
            const __bf16* Brow = Bt + (size_t)(n0 + t * 16 + ln) * K + k0;
            bfr[t] = load_frag16(Brow, half);
        }
        if (k0 + 32 < K) {                       // prefetch next K-step of A
            __builtin_prefetch(Arow0 + k0 + 32, 0, 1);
            __builtin_prefetch(Arow1 + k0 + 32, 0, 1);
        }
        // WMMA burst: 8 matrix ops, 2 independent accumulator chains per B frag
#pragma unroll
        for (int t = 0; t < 4; ++t) {
            acc0[t] = wmma_bf16(a0, bfr[t], acc0[t]);
            acc1[t] = wmma_bf16(a1, bfr[t], acc1[t]);
        }
    }
#pragma unroll
    for (int t = 0; t < 4; ++t) {
        int n = n0 + t * 16 + ln;
        float bvv = (bias && n >= bias_from) ? bias[n - bias_from] : 0.0f;
#pragma unroll
        for (int v = 0; v < 8; ++v) {
            int m = m0 + half * 8 + v;                 // C layout: M = v + 8*half
            C[(size_t)m * N + n]        = acc0[t][v] + bvv;
            C[(size_t)(m + 16) * N + n] = acc1[t][v] + bvv;
        }
    }
}

// ---------------- RoPE + head-split + V transpose ----------------
// QKV fp32 [BS][3072] -> Qb[b*H+h][s][d], Kb[b*KV+kv][s][d], Vt[b*KV+kv][d][s]  (bf16)
__global__ __launch_bounds__(256) void k_rope_split(const float* __restrict__ QKV,
                                                    const float* __restrict__ cosb,
                                                    const float* __restrict__ sinb,
                                                    __bf16* __restrict__ Qb,
                                                    __bf16* __restrict__ Kb,
                                                    __bf16* __restrict__ Vt) {
    int row = blockIdx.x;                 // b*S + s
    int b = row / Ssz, s = row % Ssz;
    const float* x  = QKV + (size_t)row * NQKV;
    const float* cs = cosb + (size_t)row * HD;
    const float* sn = sinb + (size_t)row * HD;
    for (int c = threadIdx.x; c < NQKV; c += blockDim.x) {
        float val = x[c];
        if (c < 2048) {                               // Q heads
            int h = c >> 6, d = c & 63;
            float rot = (d & 32) ? x[c - 32] : -x[c + 32];
            float r = val * cs[d] + rot * sn[d];
            Qb[((size_t)(b * Hq + h) * Ssz + s) * HD + d] = f2bf(r);
        } else if (c < 2560) {                        // K heads
            int cc = c - 2048;
            int kv = cc >> 6, d = cc & 63;
            float rot = (d & 32) ? x[c - 32] : -x[c + 32];
            float r = val * cs[d] + rot * sn[d];
            Kb[((size_t)(b * KVh + kv) * Ssz + s) * HD + d] = f2bf(r);
        } else {                                      // V heads (bias already added)
            int cc = c - 2560;
            int kv = cc >> 6, d = cc & 63;
            Vt[((size_t)(b * KVh + kv) * HD + d) * Ssz + s] = f2bf(val);
        }
    }
}

// ---------------- flash attention: one wave per (b, h, 16-query tile) ----------------
__global__ __launch_bounds__(32) void k_flash(const __bf16* __restrict__ Qb,
                                              const __bf16* __restrict__ Kb,
                                              const __bf16* __restrict__ Vt,
                                              const int* __restrict__ amask,
                                              __bf16* __restrict__ Ob) {
    __shared__ __bf16 Pt[16][32];         // probs bounce tile: C-layout -> A-layout
    const int lane = threadIdx.x & 31;
    const int half = lane >> 4;
    const int ln   = lane & 15;
    const int qt = blockIdx.x, h = blockIdx.y, b = blockIdx.z;
    const int kvh = h / (Hq / KVh);
    const int q0 = qt * 16;

    // Q fragments (A operand): row = q0 + ln, K dim = head dim, two 32-wide steps
    const __bf16* Qrow = Qb + ((size_t)(b * Hq + h) * Ssz + q0 + ln) * HD;
    v16bf aq0 = load_frag16(Qrow, half);
    v16bf aq1 = load_frag16(Qrow + 32, half);

    const __bf16* Kbase = Kb + (size_t)(b * KVh + kvh) * Ssz * HD;
    const __bf16* Vbase = Vt + (size_t)(b * KVh + kvh) * HD * Ssz;

    v8f o[4] = {};
    float mr[8], lr[8], alpha[8];
#pragma unroll
    for (int v = 0; v < 8; ++v) { mr[v] = NEG_BIG; lr[v] = 0.0f; }
    const float scale = 0.125f;           // 1/sqrt(64)

    for (int kc = 0; kc <= q0 + 15; kc += 32) {
        // ---- scores: S[16 q][32 keys] via 4 WMMAs ----
        float p[2][8];
        float cmax[8];
#pragma unroll
        for (int v = 0; v < 8; ++v) cmax[v] = NEG_BIG;
#pragma unroll
        for (int j = 0; j < 2; ++j) {
            int key = kc + j * 16 + ln;
            const __bf16* Krow = Kbase + (size_t)key * HD;
            v16bf bk0 = load_frag16(Krow, half);
            v16bf bk1 = load_frag16(Krow + 32, half);
            v8f sc = {};
            sc = wmma_bf16(aq0, bk0, sc);
            sc = wmma_bf16(aq1, bk1, sc);
            bool kv_ok = (amask[b * Ssz + key] != 0);
#pragma unroll
            for (int v = 0; v < 8; ++v) {
                int qpos = q0 + half * 8 + v;
                float sv = sc[v] * scale;
                if (!kv_ok || key > qpos) sv = NEG_BIG;
                p[j][v] = sv;
                cmax[v] = fmaxf(cmax[v], sv);
            }
        }
        // ---- row max across the 16 lanes of each half ----
#pragma unroll
        for (int v = 0; v < 8; ++v) {
#pragma unroll
            for (int off = 1; off < 16; off <<= 1)
                cmax[v] = fmaxf(cmax[v], __shfl_xor(cmax[v], off, 32));
        }
        // ---- online softmax update ----
        float rsum[8];
#pragma unroll
        for (int v = 0; v < 8; ++v) {
            float mnew = fmaxf(mr[v], cmax[v]);
            alpha[v] = __expf(mr[v] - mnew);
            mr[v] = mnew;
            float s0 = (p[0][v] <= NEG_BIG) ? 0.0f : __expf(p[0][v] - mnew);
            float s1 = (p[1][v] <= NEG_BIG) ? 0.0f : __expf(p[1][v] - mnew);
            p[0][v] = s0; p[1][v] = s1;
            rsum[v] = s0 + s1;
        }
#pragma unroll
        for (int v = 0; v < 8; ++v) {
#pragma unroll
            for (int off = 1; off < 16; off <<= 1)
                rsum[v] += __shfl_xor(rsum[v], off, 32);
            lr[v] = lr[v] * alpha[v] + rsum[v];
        }
        // ---- write P (C-layout) to LDS, reload as A fragment ----
#pragma unroll
        for (int j = 0; j < 2; ++j)
#pragma unroll
            for (int v = 0; v < 8; ++v)
                Pt[half * 8 + v][j * 16 + ln] = f2bf(p[j][v]);
        asm volatile("s_wait_dscnt 0x0" ::: "memory");
        v16bf ap;
#pragma unroll
        for (int e = 0; e < 16; ++e) {
            int k = ((e & 8) ? 16 : 0) + half * 8 + (e & 7);
            ap[e] = Pt[ln][k];
        }
        // ---- rescale O, load all 4 V fragments, then accumulate P*V ----
        v16bf vfr[4];
#pragma unroll
        for (int t = 0; t < 4; ++t) {
            const __bf16* Vrow = Vbase + (size_t)(t * 16 + ln) * Ssz + kc;
            vfr[t] = load_frag16(Vrow, half);
        }
#pragma unroll
        for (int t = 0; t < 4; ++t) {
#pragma unroll
            for (int v = 0; v < 8; ++v) o[t][v] *= alpha[v];
            o[t] = wmma_bf16(ap, vfr[t], o[t]);
        }
        asm volatile("s_wait_dscnt 0x0" ::: "memory");
    }
    // ---- epilogue: normalize, store bf16 O[b][s][h*HD + d] ----
#pragma unroll
    for (int t = 0; t < 4; ++t) {
#pragma unroll
        for (int v = 0; v < 8; ++v) {
            int qpos = q0 + half * 8 + v;
            float ov = (lr[v] > 0.0f) ? (o[t][v] / lr[v]) : 0.0f;
            Ob[((size_t)(b * Ssz + qpos)) * HHD + h * HD + t * 16 + ln] = f2bf(ov);
        }
    }
}

// ---------------- host-side launch ----------------
extern "C" void kernel_launch(void* const* d_in, const int* in_sizes, int n_in,
                              void* d_out, int out_size, void* d_ws, size_t ws_size,
                              hipStream_t stream) {
    const float* hs    = (const float*)d_in[0];
    const int*   amask = (const int*)  d_in[1];
    const float* cosb  = (const float*)d_in[2];
    const float* sinb  = (const float*)d_in[3];
    const float* Wq    = (const float*)d_in[4];
    const float* Wk    = (const float*)d_in[5];
    const float* Wv    = (const float*)d_in[6];
    const float* bv    = (const float*)d_in[7];
    const float* Wo    = (const float*)d_in[8];
    const float* bo    = (const float*)d_in[9];
    float* out = (float*)d_out;

    // workspace carve (bytes)
    char* w = (char*)d_ws;
    __bf16* Xb    = (__bf16*)w; w += (size_t)BS * DM * 2;          // 16 MB
    __bf16* Wqkvt = (__bf16*)w; w += (size_t)NQKV * DM * 2;        // 12 MB
    __bf16* Wot   = (__bf16*)w; w += (size_t)DM * HHD * 2;         //  8 MB
    float*  QKV   = (float*) w; w += (size_t)BS * NQKV * 4;        // 48 MB
    __bf16* Qb    = (__bf16*)w; w += (size_t)Bsz * Hq  * Ssz * HD * 2;  // 16 MB
    __bf16* Kb    = (__bf16*)w; w += (size_t)Bsz * KVh * Ssz * HD * 2;  //  4 MB
    __bf16* Vtb   = (__bf16*)w; w += (size_t)Bsz * KVh * Ssz * HD * 2;  //  4 MB
    __bf16* Ob    = (__bf16*)w; w += (size_t)BS * HHD * 2;         // 16 MB

    // 1) convert activations + transpose-convert weights to bf16
    k_conv_bf16<<<(BS * DM + 255) / 256, 256, 0, stream>>>(hs, Xb, BS * DM);
    k_transpose_bf16<<<(2048 * 2048 + 255) / 256, 256, 0, stream>>>(Wq, Wqkvt, DM, 2048);
    k_transpose_bf16<<<(512 * 2048 + 255) / 256, 256, 0, stream>>>(Wk, Wqkvt + (size_t)2048 * DM, DM, 512);
    k_transpose_bf16<<<(512 * 2048 + 255) / 256, 256, 0, stream>>>(Wv, Wqkvt + (size_t)2560 * DM, DM, 512);
    k_transpose_bf16<<<(2048 * 2048 + 255) / 256, 256, 0, stream>>>(Wo, Wot, HHD, DM);

    // 2) fused QKV projection (bias only on V columns, cols >= 2560)
    k_gemm_bf16<<<dim3(NQKV / 64, BS / 256), 256, 0, stream>>>(
        Xb, Wqkvt, QKV, bv, BS, NQKV, DM, 2560);

    // 3) RoPE + head split + V transpose
    k_rope_split<<<dim3(BS), 256, 0, stream>>>(QKV, cosb, sinb, Qb, Kb, Vtb);

    // 4) causal GQA flash attention
    k_flash<<<dim3(Ssz / 16, Hq, Bsz), 32, 0, stream>>>(Qb, Kb, Vtb, amask, Ob);

    // 5) output projection
    k_gemm_bf16<<<dim3(DM / 64, BS / 256), 256, 0, stream>>>(
        Ob, Wot, out, bo, BS, DM, HHD, 0);
}
